// RoomBoundaryCasting_9320079032889
// MI455X (gfx1250) — compile-verified, matrix operand order checked by reading, not run on hardware
//
#include <hip/hip_runtime.h>
#include <hip/hip_bf16.h>

typedef __attribute__((ext_vector_type(16))) _Float16 v16h;
typedef __attribute__((ext_vector_type(8)))  float    v8f;
typedef __attribute__((ext_vector_type(4)))  float    v4f;

#define VOX 128
#define NBOX 16
#define XCHUNKS 8   // 128 / 16 rows per block
#define ZCHUNKS 4   // z split: 32 z-values per block

// Grid: NBOX * XCHUNKS * ZCHUNKS blocks, 256 threads (8 wave32) per block.
// Block (b, ic, zc):
//   phase 1: build occ_x/occ_y/occ_z (exact replication of reference rounding) in LDS
//   phase 2: 8 waves compute 16x16 outer-product tiles of occ_x(chunk) x occ_y via
//            v_wmma_f32_16x16x32_f16 into an LDS plane[16][128]
//   phase 3: stream plane * occ_z as v4f non-temporal B128 stores (coalesced in z)
__global__ __launch_bounds__(256) void room_boundary_kernel(
    const float* __restrict__ bbox,   // [NBOX, 6] = (max_x,max_y,max_z,min_x,min_y,min_z)
    float* __restrict__ out)          // [NBOX, 128, 128, 128]
{
    __shared__ float occ[3][VOX];       // per-axis occupancy (0/1)
    __shared__ float plane[16 * VOX];   // occ_x-chunk (x) occ_y outer product

    const int tid  = threadIdx.x;
    const int blk  = blockIdx.x;
    const int zc   = blk & (ZCHUNKS - 1);
    const int ic   = (blk >> 2) & (XCHUNKS - 1);
    const int b    = blk >> 5;
    const int i0   = ic * 16;
    const int z0   = zc * (VOX / ZCHUNKS);   // 32 z values

    // ---- phase 1: occupancy vectors ------------------------------------
    if (tid < VOX) {
        occ[0][tid] = 0.0f;
        occ[1][tid] = 0.0f;
        occ[2][tid] = 0.0f;
    }
    __syncthreads();

    if (tid < VOX) {
        const float* bb = bbox + b * 6;
        const float x = (float)tid;
        #pragma unroll
        for (int d = 0; d < 3; ++d) {
            const float mn = bb[3 + d];
            const float mx = bb[d];
            // scale = (max-min)/128 : exact (power-of-two divide)
            const float sc = __fmul_rn(__fsub_rn(mx, mn), (1.0f / 128.0f));
            // reference: (grid * scale + min).astype(int32)  -- mul, add, trunc
            int idx = (int)__fadd_rn(__fmul_rn(x, sc), mn);
            idx = idx < 0 ? 0 : (idx > VOX - 1 ? VOX - 1 : idx);
            occ[d][idx] = 1.0f;   // benign write race: all writers store 1.0
        }
    }
    __syncthreads();

    // ---- phase 2: rank-1 WMMA tiles: plane[i][j] = occx[i0+i]*occy[j] ----
    {
        const int lane = tid & 31;
        const int wave = tid >> 5;          // 8 waves -> j tile = wave*16
        const int j0w  = wave * 16;
        const int li   = lane & 15;

        const float axf = occ[0][i0 + li];
        const float byf = occ[1][j0w + li];

        v16h a = {};
        v16h bv = {};
        // 16-bit A (16x32): lanes 0-15 hold K=0,1 in VGPR0 -> element 0 = K=0 col.
        // B (32x16): lanes 0-15 hold K=0.. -> element 0 = K=0 row. All other K = 0.
        a[0]  = (lane < 16) ? (_Float16)axf : (_Float16)0.0f;
        bv[0] = (lane < 16) ? (_Float16)byf : (_Float16)0.0f;

        v8f c = {};
        // D[m][n] = sum_k A[m][k]*B[k][n] = occx[m] * occy[n]
        c = __builtin_amdgcn_wmma_f32_16x16x32_f16(
                /*neg_a=*/false, a, /*neg_b=*/false, bv,
                /*c_mod=*/(short)0, c, /*reuse_a=*/false, /*reuse_b=*/false);

        // C/D layout: lane l, VGPR r -> M = r + 8*(l>=16), N = l&15
        const int n  = lane & 15;
        const int hi = lane >> 4;
        #pragma unroll
        for (int r = 0; r < 8; ++r)
            plane[(r + 8 * hi) * VOX + j0w + n] = c[r];
    }
    __syncthreads();

    // ---- phase 3: stream plane x occ_z to global (v4f NT stores) --------
    // Per block: 16 rows * 128 j * 8 v4f(z) = 16384 v4fs; 64 per thread.
    const size_t base = ((size_t)b * VOX + (size_t)i0) * VOX * VOX;
    const v4f* occz4 = (const v4f*)&occ[2][z0];

    #pragma unroll 4
    for (int it = 0; it < 64; ++it) {
        const int linear = it * 256 + tid;
        const int zq = linear & 7;            // v4f index within 32-z window
        const int j  = (linear >> 3) & (VOX - 1);
        const int i  = linear >> 10;          // 0..15

        const float v  = plane[i * VOX + j];
        const v4f   oz = occz4[zq];
        const v4f   o  = v * oz;

        v4f* dst = (v4f*)(out + base + ((size_t)i * VOX + j) * VOX
                          + (size_t)z0 + (size_t)zq * 4);
        __builtin_nontemporal_store(o, dst);
    }
}

extern "C" void kernel_launch(void* const* d_in, const int* in_sizes, int n_in,
                              void* d_out, int out_size, void* d_ws, size_t ws_size,
                              hipStream_t stream) {
    (void)in_sizes; (void)n_in; (void)d_ws; (void)ws_size; (void)out_size;
    const float* bbox = (const float*)d_in[0];
    float* out = (float*)d_out;
    dim3 grid(NBOX * XCHUNKS * ZCHUNKS);   // 512 blocks
    dim3 block(256);                       // 8 wave32
    room_boundary_kernel<<<grid, block, 0, stream>>>(bbox, out);
}
